// MultiheadAttentionMechanism_42408507081022
// MI455X (gfx1250) — matrix operand order repaired and checked by reading, not compiled, not run on hardware
//
#include <hip/hip_runtime.h>

// ---------------------------------------------------------------------------
// Multi-head attention for MI455X (gfx1250).
// Matrix math: split-bf16 (bf16x2 error-compensated) on V_WMMA_F32_16X16X32_BF16
//   x = hi + lo,  x*y ~= hi*hi + hi*lo + lo*hi   (fp32 accumulate, ~1e-5 of fp32)
// B=16, H=8, QLEN=512, KLEN=1024, D_K=64, ADIM=KDIM=512, scale = 1/8.
// d_out = cv [16,512,512] fp32 || aw [16,8,512,1024] fp32.
// ---------------------------------------------------------------------------

typedef __attribute__((ext_vector_type(4)))  float  v4f;
typedef __attribute__((ext_vector_type(8)))  float  v8f;
typedef __attribute__((ext_vector_type(8)))  __bf16 v8bf;
typedef __attribute__((ext_vector_type(16))) __bf16 v16bf;

#define NEG_INF_F (-3.4028234663852886e+38f)

__device__ __forceinline__ v8f wmma_bf16(v16bf a, v16bf b, v8f c) {
  // (neg_a, A, neg_b, B, c_mod, C, reuse_a, reuse_b)
  return __builtin_amdgcn_wmma_f32_16x16x32_bf16(false, a, false, b,
                                                 (short)0, c, false, false);
}

// B-fragment: elements 0-7 at p, 8-15 at p+16 (column N = lane&15, K = k0 + 8*(lane>>4)..)
__device__ __forceinline__ v16bf ldb16(const __bf16* p) {
  v8bf b0 = *(const v8bf*)p;
  v8bf b1 = *(const v8bf*)(p + 16);
  return __builtin_shufflevector(b0, b1, 0, 1, 2, 3, 4, 5, 6, 7, 8, 9, 10, 11,
                                 12, 13, 14, 15);
}

// A-fragment from bf16 planes: elements 0-7 at p, 8-15 at p+16 (K gap of 16)
__device__ __forceinline__ v16bf lda16_gap(const __bf16* p) {
  v8bf b0 = *(const v8bf*)p;
  v8bf b1 = *(const v8bf*)(p + 16);
  return __builtin_shufflevector(b0, b1, 0, 1, 2, 3, 4, 5, 6, 7, 8, 9, 10, 11,
                                 12, 13, 14, 15);
}

__device__ __forceinline__ void cvt4(v4f x, v16bf& h, v16bf& l, int base) {
#pragma unroll
  for (int i = 0; i < 4; ++i) {
    float xv = x[i];
    __bf16 hb = (__bf16)xv;
    h[base + i] = hb;
    l[base + i] = (__bf16)(xv - (float)hb);
  }
}

// fp32 row-major -> split-bf16 A fragment for one K=32 chunk.
// Element i holds K = k0+hi8+i (i<8) or K = k0+16+hi8+(i-8)  (ISA 7.12.2).
// xp must point at  row_base + k0 + (lanehi<<3).
__device__ __forceinline__ void load_a_split(const float* xp, v16bf& h,
                                             v16bf& l) {
  cvt4(*(const v4f*)(xp), h, l, 0);
  cvt4(*(const v4f*)(xp + 4), h, l, 4);
  cvt4(*(const v4f*)(xp + 16), h, l, 8);
  cvt4(*(const v4f*)(xp + 20), h, l, 12);
}

// ---------------------------------------------------------------------------
// Split an fp32 array into (hi, lo) bf16 planes (for the reused B operands).
// ---------------------------------------------------------------------------
__global__ __launch_bounds__(256) void split_planes(
    const float* __restrict__ X, __bf16* __restrict__ H,
    __bf16* __restrict__ L, int n) {
  int i = blockIdx.x * 256 + threadIdx.x;
  if (i < n) {
    float x = X[i];
    __bf16 hb = (__bf16)x;
    H[i] = hb;
    L[i] = (__bf16)(x - (float)hb);
  }
}

// ---------------------------------------------------------------------------
// OUT = X(M x 512) @ W(512 x 512)^T + bias, split-bf16 WMMA, fp32 accum.
//   mode 0: OUTa fp32 row-major [M,512]
//   mode 1: OUTa fp32 head-split  [B,8,L,64]
//   mode 2: OUTa/OUTb bf16 hi/lo  [B,8,L,64]
//   mode 3: OUTa/OUTb bf16 hi/lo  [B,8,64,L]   (transposed, for V)
// Block = 16 output rows; 8 waves x 64 columns.  lshift = log2(L).
// ---------------------------------------------------------------------------
__global__ __launch_bounds__(256) void gemm512_split(
    const float* __restrict__ X, const __bf16* __restrict__ Whi,
    const __bf16* __restrict__ Wlo, const float* __restrict__ bias,
    void* __restrict__ OUTa, void* __restrict__ OUTb, int mode, int lshift) {
  const int m0 = blockIdx.x << 4;
  const int lane = threadIdx.x & 31;
  const int l16 = lane & 15;
  const int hi = lane >> 4;
  const int n0 = (threadIdx.x >> 5) << 6;

  const float* xrow = X + (size_t)(m0 + l16) * 512 + (hi << 3);
  const size_t wofs = (size_t)(n0 + l16) * 512 + (hi << 3);

  v8f acc[4] = {};
  for (int k0 = 0; k0 < 512; k0 += 32) {
    __builtin_prefetch(xrow + k0 + 64, 0, 0);  // global_prefetch_b8
    v16bf ahi, alo;
    load_a_split(xrow + k0, ahi, alo);
#pragma unroll
    for (int t = 0; t < 4; ++t) {
      const size_t o = wofs + ((size_t)(t << 4) << 9) + k0;
      v16bf bhi = ldb16(Whi + o);
      v16bf blo = ldb16(Wlo + o);
      acc[t] = wmma_bf16(ahi, bhi, acc[t]);
      acc[t] = wmma_bf16(alo, bhi, acc[t]);
      acc[t] = wmma_bf16(ahi, blo, acc[t]);
    }
  }

  const int Lm1 = (1 << lshift) - 1;
#pragma unroll
  for (int t = 0; t < 4; ++t) {
    const int ncol = n0 + (t << 4) + l16;
    const float bv = bias[ncol];
    const int hh = ncol >> 6;
    const int dk = ncol & 63;
#pragma unroll
    for (int r = 0; r < 8; ++r) {
      const int row = m0 + r + (hi << 3);
      const float val = acc[t][r] + bv;
      if (mode == 0) {
        ((float*)OUTa)[(size_t)row * 512 + ncol] = val;
      } else {
        const int bb = row >> lshift;
        const int ll = row & Lm1;
        const size_t bh = (size_t)(bb * 8 + hh);
        if (mode == 1) {
          ((float*)OUTa)[((bh << lshift) + ll) * 64 + dk] = val;
        } else if (mode == 2) {
          const size_t idx = ((bh << lshift) + ll) * 64 + dk;
          __bf16 hb = (__bf16)val;
          ((__bf16*)OUTa)[idx] = hb;
          ((__bf16*)OUTb)[idx] = (__bf16)(val - (float)hb);
        } else {
          const size_t idxT = ((bh * 64 + dk) << lshift) + ll;
          __bf16 hb = (__bf16)val;
          ((__bf16*)OUTa)[idxT] = hb;
          ((__bf16*)OUTb)[idxT] = (__bf16)(val - (float)hb);
        }
      }
    }
  }
}

// ---------------------------------------------------------------------------
// Attention: one block (4 waves) per (b, h, 16-row q tile).
//  phase 1: scores = (Q K^T)/8 + mask -> LDS fp32 (waves split KLEN in 4)
//  phase 2: softmax (8 lanes/row, shfl reductions); store fp32 aw to global
//           and split aw into bf16 hi/lo planes in LDS
//  phase 3: ctx = AW @ V via split-bf16 WMMA (A from LDS, B from vT planes)
// ---------------------------------------------------------------------------
__global__ __launch_bounds__(128) void attention_kernel(
    const float* __restrict__ qf, const __bf16* __restrict__ khi,
    const __bf16* __restrict__ klo, const __bf16* __restrict__ vThi,
    const __bf16* __restrict__ vTlo, const int* __restrict__ mask,
    float* __restrict__ aw_all, float* __restrict__ ctx_all) {
  __shared__ union {
    float sS[16 * 1026];  // fp32 scores (stride 1026: aligned b64)  65,664 B
    float sC[4 * 1024];   // phase-3 partials (sS is dead by then)
  } u;
  __shared__ __bf16 awh[16 * 1032];  // stride 1032: 16B-aligned, bank-rotated
  __shared__ __bf16 awl[16 * 1032];

  const int tile = blockIdx.x;  // b*256 + h*32 + qt
  const int qt = tile & 31, h = (tile >> 5) & 7, b = tile >> 8;
  const int q0 = qt << 4;
  const int tid = threadIdx.x, wave = tid >> 5, lane = tid & 31;
  const int l16 = lane & 15, hi = lane >> 4;

  const float* qp = qf + (((size_t)(b * 8 + h)) * 512 + q0) * 64;
  const __bf16* kb = khi + ((size_t)(b * 8 + h)) * 1024 * 64;
  const __bf16* vb = vThi + ((size_t)(b * 8 + h)) * 64 * 1024;
  const ptrdiff_t klo_d = klo - khi;
  const ptrdiff_t vlo_d = vTlo - vThi;
  const int* mkp = mask + ((size_t)b * 512 + q0) * 1024;
  float* awp = aw_all + (((size_t)(b * 8 + h)) * 512 + q0) * 1024;

  // Q fragments (fp32 -> split-bf16), two K=32 chunks cover d_k = 64
  v16bf qh[2], ql[2];
  {
    const float* qr = qp + l16 * 64 + (hi << 3);
#pragma unroll
    for (int c = 0; c < 2; ++c) load_a_split(qr + (c << 5), qh[c], ql[c]);
  }

  // --- phase 1: scores for this wave's 256-column chunk ---
  for (int nt = 0; nt < 16; ++nt) {
    const int kc0 = (wave << 8) + (nt << 4);
    const __bf16* kp = kb + (size_t)(kc0 + l16) * 64 + (hi << 3);
    __builtin_prefetch(kp + 16 * 64, 0, 0);
    v8f acc = {};
#pragma unroll
    for (int c = 0; c < 2; ++c) {
      v16bf bh = ldb16(kp + (c << 5));
      v16bf bl = ldb16(kp + (c << 5) + klo_d);
      acc = wmma_bf16(qh[c], bh, acc);
      acc = wmma_bf16(ql[c], bh, acc);
      acc = wmma_bf16(qh[c], bl, acc);
    }
    const int col = kc0 + l16;
#pragma unroll
    for (int r = 0; r < 8; ++r) {
      const int row = r + (hi << 3);
      float e = acc[r] * 0.125f;  // 1/sqrt(64)
      if (mkp[row * 1024 + col] == 0) e = NEG_INF_F;
      u.sS[row * 1026 + col] = e;
    }
  }
  __syncthreads();

  // --- phase 2: softmax, 8 lanes per row (aligned lane groups of 8) ---
  {
    const int row = tid >> 3;
    const int sub = tid & 7;
    float* srow = u.sS + row * 1026;
    const int c0 = sub << 7;

    float mx = NEG_INF_F;
    for (int c = c0; c < c0 + 128; ++c) mx = fmaxf(mx, srow[c]);
    mx = fmaxf(mx, __shfl_xor(mx, 1));
    mx = fmaxf(mx, __shfl_xor(mx, 2));
    mx = fmaxf(mx, __shfl_xor(mx, 4));

    float sum = 0.f;
    for (int c = c0; c < c0 + 128; ++c) {
      float ex = expf(srow[c] - mx);
      srow[c] = ex;
      sum += ex;
    }
    sum += __shfl_xor(sum, 1);
    sum += __shfl_xor(sum, 2);
    sum += __shfl_xor(sum, 4);

    const float rs = 1.0f / sum;
    for (int c = c0; c < c0 + 128; ++c) srow[c] *= rs;
  }
  __syncthreads();

  // coalesced fp32 aw store (the 256 MB traffic floor) + split into LDS planes
  for (int idx = tid; idx < 16 * 1024; idx += 128) {
    const int r = idx >> 10, c = idx & 1023;
    const float w = u.sS[r * 1026 + c];
    awp[idx] = w;
    __bf16 hb = (__bf16)w;
    awh[r * 1032 + c] = hb;
    awl[r * 1032 + c] = (__bf16)(w - (float)hb);
  }
  __syncthreads();  // after this sS is dead -> sC may alias it

  // --- phase 3: ctx partial = AW(16 x 256) @ V(256 x 64), split-bf16 ---
  v8f cacc[4] = {};
  const int kbase = wave << 8;
  for (int kk = kbase; kk < kbase + 256; kk += 32) {
    const __bf16* ah = awh + l16 * 1032 + kk + (hi << 3);
    const __bf16* al = awl + l16 * 1032 + kk + (hi << 3);
    v16bf ahv = lda16_gap(ah);  // DS b128 pair
    v16bf alv = lda16_gap(al);
#pragma unroll
    for (int t = 0; t < 4; ++t) {
      const __bf16* vp = vb + (size_t)((t << 4) + l16) * 1024 + kk + (hi << 3);
      v16bf bh = ldb16(vp);
      v16bf bl = ldb16(vp + vlo_d);
      cacc[t] = wmma_bf16(ahv, bh, cacc[t]);
      cacc[t] = wmma_bf16(alv, bh, cacc[t]);
      cacc[t] = wmma_bf16(ahv, bl, cacc[t]);
    }
  }
  {
    float* myC = u.sC + (wave << 10);
#pragma unroll
    for (int t = 0; t < 4; ++t)
#pragma unroll
      for (int r = 0; r < 8; ++r)
        myC[(r + (hi << 3)) * 64 + (t << 4) + l16] = cacc[t][r];
  }
  __syncthreads();

  float* ctxp = ctx_all + ((size_t)b * 512 + q0) * 512 + h * 64;
  for (int idx = tid; idx < 1024; idx += 128) {
    const int row = idx >> 6;
    const int col = idx & 63;
    const float s =
        u.sC[idx] + u.sC[1024 + idx] + u.sC[2048 + idx] + u.sC[3072 + idx];
    ctxp[(size_t)row * 512 + col] = s;
  }
}

// ---------------------------------------------------------------------------
extern "C" void kernel_launch(void* const* d_in, const int* in_sizes, int n_in,
                              void* d_out, int out_size, void* d_ws,
                              size_t ws_size, hipStream_t stream) {
  (void)in_sizes; (void)n_in; (void)out_size; (void)ws_size;

  const float* key = (const float*)d_in[0];    // [16,1024,512]
  const float* value = (const float*)d_in[1];  // [16,1024,512]
  const float* query = (const float*)d_in[2];  // [16,512,512]
  const int* mask = (const int*)d_in[3];       // [16,512,1024]
  const float* Wk = (const float*)d_in[4];
  const float* bk = (const float*)d_in[5];
  const float* Wv = (const float*)d_in[6];
  const float* bv = (const float*)d_in[7];
  const float* Wq = (const float*)d_in[8];
  const float* bq = (const float*)d_in[9];
  const float* Wo = (const float*)d_in[10];
  const float* bo = (const float*)d_in[11];

  float* out = (float*)d_out;
  float* cv_out = out;            // [16,512,512] fp32
  float* aw_out = out + 4194304;  // [16,8,512,1024] fp32

  // workspace carve-up (~105 MB total)
  char* w = (char*)d_ws;
  auto take = [&](size_t bytes) {
    char* p = w;
    w += (bytes + 255) & ~(size_t)255;
    return p;
  };
  const size_t WPLANE = (size_t)512 * 512 * 2;        // one W bf16 plane
  __bf16* Wkh = (__bf16*)take(WPLANE);
  __bf16* Wkl = (__bf16*)take(WPLANE);
  __bf16* Wvh = (__bf16*)take(WPLANE);
  __bf16* Wvl = (__bf16*)take(WPLANE);
  __bf16* Wqh = (__bf16*)take(WPLANE);
  __bf16* Wql = (__bf16*)take(WPLANE);
  __bf16* Woh = (__bf16*)take(WPLANE);
  __bf16* Wol = (__bf16*)take(WPLANE);
  float* qb = (float*)take((size_t)16 * 8 * 512 * 64 * 4);    // fp32 q
  __bf16* kh = (__bf16*)take((size_t)16 * 8 * 1024 * 64 * 2); // k hi plane
  __bf16* kl = (__bf16*)take((size_t)16 * 8 * 1024 * 64 * 2);
  __bf16* vTh = (__bf16*)take((size_t)16 * 8 * 64 * 1024 * 2);// v^T hi plane
  __bf16* vTl = (__bf16*)take((size_t)16 * 8 * 64 * 1024 * 2);
  float* ctx = (float*)take((size_t)16 * 512 * 512 * 4);      // fp32 ctx

  // split the reused B operands (weights) into bf16 hi/lo planes
  const int WN = 512 * 512;
  split_planes<<<WN / 256, 256, 0, stream>>>(Wk, Wkh, Wkl, WN);
  split_planes<<<WN / 256, 256, 0, stream>>>(Wv, Wvh, Wvl, WN);
  split_planes<<<WN / 256, 256, 0, stream>>>(Wq, Wqh, Wql, WN);
  split_planes<<<WN / 256, 256, 0, stream>>>(Wo, Woh, Wol, WN);

  // projections (A converted on the fly; k/v emitted as split planes)
  gemm512_split<<<1024, 256, 0, stream>>>(key, Wkh, Wkl, bk, kh, kl, 2, 10);
  gemm512_split<<<1024, 256, 0, stream>>>(value, Wvh, Wvl, bv, vTh, vTl, 3, 10);
  gemm512_split<<<512, 256, 0, stream>>>(query, Wqh, Wql, bq, qb, nullptr, 1, 9);

  // attention core: scores -> softmax -> aw store -> ctx
  attention_kernel<<<4096, 128, 0, stream>>>(qb, kh, kl, vTh, vTl, mask,
                                             aw_out, ctx);

  // output projection (fp32 result)
  gemm512_split<<<512, 256, 0, stream>>>(ctx, Woh, Wol, bo, cv_out, nullptr, 0,
                                         9);
}
